// EncoderBlock_11098195493325
// MI455X (gfx1250) — matrix operand order repaired and checked by reading, not compiled, not run on hardware
//
#include <hip/hip_runtime.h>
#include <hip/hip_bf16.h>
#include <math.h>

// ---------------------------------------------------------------------------
// Types for CDNA5 WMMA (wave32, 16x16x32 bf16 -> f32)
// ---------------------------------------------------------------------------
typedef __attribute__((ext_vector_type(16))) __bf16 bf16x16;
typedef __attribute__((ext_vector_type(8)))  float  f32x8;
typedef __attribute__((ext_vector_type(4)))  float  f32x4;

union Frag {
    bf16x16 v;
    f32x4   f4[2];
};

__device__ __forceinline__ unsigned short f32_to_bf16(float f) {
    unsigned int u = __float_as_uint(f);
    u += 0x7FFFu + ((u >> 16) & 1u);      // round-to-nearest-even
    return (unsigned short)(u >> 16);
}

__device__ __forceinline__ f32x8 wmma_bf16(const Frag& a, const Frag& b, f32x8 c) {
    return __builtin_amdgcn_wmma_f32_16x16x32_bf16(
        /*neg_a=*/false, a.v, /*neg_b=*/false, b.v,
        /*c_mod=*/(short)0, c, /*reuse_a=*/false, /*reuse_b=*/false);
}

// ---------------------------------------------------------------------------
// Tensor Data Mover (TDM) support: async 2D tile DMA global -> LDS.
// Guarded: falls back to vector loads if the builtin is unavailable.
// ---------------------------------------------------------------------------
#if __has_builtin(__builtin_amdgcn_tensor_load_to_lds) && __has_builtin(__builtin_amdgcn_s_wait_tensorcnt)
#define USE_TDM 1
#else
#define USE_TDM 0
#endif

typedef __attribute__((ext_vector_type(4))) unsigned tdm_u32x4;
typedef __attribute__((ext_vector_type(8))) int      tdm_i32x8;
typedef __attribute__((ext_vector_type(4))) int      tdm_i32x4;

// 2D bf16 tile load: tile_elems_x elements per row, tile_rows rows,
// global row stride row_stride_elems. LDS rows padded 64B->80B by the TDM
// (pad_interval = 16 DWORDs, pad_amount = 4 DWORDs) to match our LDS pitch.
__device__ __forceinline__ void tdm_load_tile_2d(unsigned lds_byte_addr,
                                                 const void* gptr,
                                                 unsigned tile_elems_x,
                                                 unsigned tile_rows,
                                                 unsigned row_stride_elems) {
#if USE_TDM
    unsigned long long ga = (unsigned long long)gptr;
    tdm_u32x4 g0;
    g0[0] = 1u;                                                   // count=1 (valid), user mode
    g0[1] = lds_byte_addr;                                        // lds_addr
    g0[2] = (unsigned)ga;                                         // global_addr[31:0]
    g0[3] = (unsigned)((ga >> 32) & 0x01FFFFFFu) | 0x80000000u;   // global_addr[56:32] | type=2
    unsigned td0 = row_stride_elems;                              // tensor_dim0 (row length)
    unsigned td1 = tile_rows;                                     // tensor_dim1 (rows from tile start)
    tdm_i32x8 g1;
    g1[0] = (int)((1u << 16)        // data_size = 2 bytes
                | (1u << 20)        // pad_enable
                | (3u << 22)        // pad_interval: 16 DWORDs (64B)
                | (3u << 25));      // pad_amount:   4 DWORDs (16B)
    g1[1] = (int)((td0 & 0xFFFFu) << 16);                                   // tensor_dim0 lo
    g1[2] = (int)(((td0 >> 16) & 0xFFFFu) | ((td1 & 0xFFFFu) << 16));       // dim0 hi | dim1 lo
    g1[3] = (int)(((td1 >> 16) & 0xFFFFu) | ((tile_elems_x & 0xFFFFu) << 16)); // dim1 hi | tile_dim0
    g1[4] = (int)(tile_rows & 0xFFFFu);                                     // tile_dim1, tile_dim2=0
    g1[5] = (int)row_stride_elems;                                          // tensor_dim0_stride lo
    g1[6] = 0;
    g1[7] = 0;
    tdm_i32x4 z4 = {0, 0, 0, 0};
#if defined(__clang_major__) && (__clang_major__ >= 23)
    tdm_i32x8 z8 = {0, 0, 0, 0, 0, 0, 0, 0};
    __builtin_amdgcn_tensor_load_to_lds(g0, g1, z4, z4, z8, 0);
#else
    __builtin_amdgcn_tensor_load_to_lds(g0, g1, z4, z4, 0);
#endif
#else
    (void)lds_byte_addr; (void)gptr; (void)tile_elems_x; (void)tile_rows; (void)row_stride_elems;
#endif
}

__device__ __forceinline__ unsigned lds_byte_addr_of(const void* p) {
    return (unsigned)(unsigned long long)p;   // low 32 bits of flat addr = LDS offset
}

// ---------------------------------------------------------------------------
// f32 -> bf16 conversion (weights)
// ---------------------------------------------------------------------------
__global__ __launch_bounds__(256) void cvt_bf16_kernel(
    const float* __restrict__ src, unsigned short* __restrict__ dst, int n) {
    int i = blockIdx.x * blockDim.x + threadIdx.x;
    int stride = gridDim.x * blockDim.x;
    for (; i < n; i += stride) dst[i] = f32_to_bf16(src[i]);
}

// ---------------------------------------------------------------------------
// LayerNorm: one block (256 thr) per row of 1024. Unbiased std (ddof=1), no eps.
// ---------------------------------------------------------------------------
template<bool OUTF, bool OUTB>
__global__ __launch_bounds__(256) void ln_kernel(
    const float* __restrict__ x,
    const float* __restrict__ alpha, const float* __restrict__ beta,
    float* __restrict__ outf, unsigned short* __restrict__ outb) {
    const int D = 1024;
    int row = blockIdx.x;
    const float* xr = x + (size_t)row * D;
    int t = threadIdx.x;
    float vals[4];
    float s = 0.f, s2 = 0.f;
#pragma unroll
    for (int i = 0; i < 4; i++) {
        float v = xr[t + i * 256];
        vals[i] = v; s += v; s2 += v * v;
    }
#pragma unroll
    for (int off = 16; off > 0; off >>= 1) {
        s  += __shfl_down(s,  off);
        s2 += __shfl_down(s2, off);
    }
    __shared__ float red[2][8];
    int w = t >> 5;
    if ((t & 31) == 0) { red[0][w] = s; red[1][w] = s2; }
    __syncthreads();
    s = 0.f; s2 = 0.f;
#pragma unroll
    for (int i = 0; i < 8; i++) { s += red[0][i]; s2 += red[1][i]; }
    float mean = s * (1.0f / 1024.0f);
    float var  = (s2 - 1024.0f * mean * mean) * (1.0f / 1023.0f);
    float rstd = rsqrtf(var);
#pragma unroll
    for (int i = 0; i < 4; i++) {
        int c = t + i * 256;
        float y = (vals[i] - mean) * rstd * alpha[c] + beta[c];
        if (OUTF) outf[(size_t)row * D + c] = y;
        if (OUTB) outb[(size_t)row * D + c] = f32_to_bf16(y);
    }
}

// ---------------------------------------------------------------------------
// WMMA GEMM: C[M,N] = act(A[M,K] * W[N,K]^T + bias) (+ residual)
// Block: 256 thr = 8 waves (4 M x 2 N), block tile 128x128, BK=32,
// wave tile 32x64 => 8 f32x8 accumulators / wave, 8 wmma per K-step
// against 12 ds_load_b128. LDS rows padded to 40 bf16 (80B = 5*16B):
// b128-aligned and bank-tiled. Tiles are DMA'd by the TDM (double-buffered,
// TENSORcnt-tracked) when available.
// ---------------------------------------------------------------------------
#define GEMM_BM 128
#define GEMM_BN 128
#define GEMM_BK 32
#define GEMM_LD 40

template<bool BIAS, bool RELU, bool RES, bool OUTF, bool OUTB>
__global__ __launch_bounds__(256) void gemm_bf16_kernel(
    const __bf16* __restrict__ A, const __bf16* __restrict__ W,
    const float* __restrict__ bias, const float* __restrict__ res,
    float* __restrict__ outf, unsigned short* __restrict__ outb,
    int M, int N, int K) {
    __shared__ __bf16 As[2][GEMM_BM * GEMM_LD];
    __shared__ __bf16 Ws[2][GEMM_BN * GEMM_LD];

    int t = threadIdx.x;
    int wave = t >> 5, lane = t & 31;
    int lo = lane & 15, hf = lane >> 4;
    int wm = wave & 3, wn = wave >> 2;          // 4 x 2 wave grid -> 32 x 64 / wave
    int m0 = blockIdx.y * GEMM_BM;
    int n0 = blockIdx.x * GEMM_BN;

    f32x8 acc[2][4];
#pragma unroll
    for (int mt = 0; mt < 2; mt++)
#pragma unroll
        for (int nt = 0; nt < 4; nt++)
#pragma unroll
            for (int r = 0; r < 8; r++) acc[mt][nt][r] = 0.f;

    const int nsteps = K / GEMM_BK;

#if USE_TDM
    if (wave == 0) {
        tdm_load_tile_2d(lds_byte_addr_of(&As[0][0]), &A[(size_t)m0 * K], GEMM_BK, GEMM_BM, K);
        tdm_load_tile_2d(lds_byte_addr_of(&Ws[0][0]), &W[(size_t)n0 * K], GEMM_BK, GEMM_BN, K);
    }
#endif

    for (int kb = 0; kb < nsteps; kb++) {
        int cur = kb & 1;
#if USE_TDM
        if (wave == 0) {
            if (kb + 1 < nsteps) {
                int k0n = (kb + 1) * GEMM_BK;
                tdm_load_tile_2d(lds_byte_addr_of(&As[cur ^ 1][0]),
                                 &A[(size_t)m0 * K + k0n], GEMM_BK, GEMM_BM, K);
                tdm_load_tile_2d(lds_byte_addr_of(&Ws[cur ^ 1][0]),
                                 &W[(size_t)n0 * K + k0n], GEMM_BK, GEMM_BN, K);
                __builtin_amdgcn_s_wait_tensorcnt(2);   // drain previous pair
            } else {
                __builtin_amdgcn_s_wait_tensorcnt(0);
            }
        }
        __syncthreads();
#else
        {
            int k0 = kb * GEMM_BK;
            // A tile: 128 rows x 32 bf16 = 512 x 16B chunks / 256 thr = 2 each
#pragma unroll
            for (int i = 0; i < 2; i++) {
                int idx = t + i * 256;
                int r = idx >> 2, sseg = (idx & 3) * 8;
                *(f32x4*)&As[cur][r * GEMM_LD + sseg] =
                    *(const f32x4*)&A[(size_t)(m0 + r) * K + k0 + sseg];
            }
            // W tile: 128 rows x 32 bf16 = 512 x 16B chunks -> 2 each
#pragma unroll
            for (int i = 0; i < 2; i++) {
                int idx = t + i * 256;
                int r = idx >> 2, sseg = (idx & 3) * 8;
                *(f32x4*)&Ws[cur][r * GEMM_LD + sseg] =
                    *(const f32x4*)&W[(size_t)(n0 + r) * K + k0 + sseg];
            }
        }
        __syncthreads();
#endif

        Frag af[2], bf[4];
#pragma unroll
        for (int mt = 0; mt < 2; mt++) {
            const __bf16* ap = &As[cur][(wm * 32 + mt * 16 + lo) * GEMM_LD + 8 * hf];
            af[mt].f4[0] = *(const f32x4*)ap;          // K = 8*hf .. +7
            af[mt].f4[1] = *(const f32x4*)(ap + 16);   // K = 16+8*hf .. +7
        }
#pragma unroll
        for (int nt = 0; nt < 4; nt++) {
            const __bf16* wp = &Ws[cur][(wn * 64 + nt * 16 + lo) * GEMM_LD + 8 * hf];
            bf[nt].f4[0] = *(const f32x4*)wp;          // K = 8*hf .. +7
            bf[nt].f4[1] = *(const f32x4*)(wp + 16);   // K = 16+8*hf .. +7
        }
#pragma unroll
        for (int mt = 0; mt < 2; mt++)
#pragma unroll
            for (int nt = 0; nt < 4; nt++)
                acc[mt][nt] = wmma_bf16(af[mt], bf[nt], acc[mt][nt]);
        __syncthreads();
    }

    int gm = m0 + wm * 32, gn = n0 + wn * 64;
#pragma unroll
    for (int mt = 0; mt < 2; mt++) {
#pragma unroll
        for (int nt = 0; nt < 4; nt++) {
            int col = gn + nt * 16 + lo;
            float bval = BIAS ? bias[col] : 0.0f;
#pragma unroll
            for (int r = 0; r < 8; r++) {
                int row = gm + mt * 16 + r + 8 * hf;
                float v = acc[mt][nt][r] + bval;
                if (RELU) v = fmaxf(v, 0.0f);
                size_t o = (size_t)row * N + col;
                if (RES) v += res[o];
                if (OUTF) outf[o] = v;
                if (OUTB) outb[o] = f32_to_bf16(v);
            }
        }
    }
}

// ---------------------------------------------------------------------------
// Flash attention: block = (b, h, 128 q-rows), 8 waves x 16 q-rows each.
// Streams 64-key tiles; online softmax; WMMA for Q*K^T and P*V.
// mask is key-only (B,1,1,S), NEG applied pre-softmax like the reference.
// ---------------------------------------------------------------------------
__global__ __launch_bounds__(256) void attn_kernel(
    const __bf16* __restrict__ Q, const __bf16* __restrict__ Kg,
    const __bf16* __restrict__ Vg, const unsigned char* __restrict__ mask,
    unsigned short* __restrict__ O) {
    const int S = 2048, DM = 1024;
    __shared__ __bf16 Ks[64 * 72];       // [key][d]
    __shared__ __bf16 Vt[64 * 72];       // [d][key]
    __shared__ __bf16 Ps[8 * 16 * 72];   // per-wave P staging [qrow][key]

    int t = threadIdx.x;
    int wave = t >> 5, lane = t & 31;
    int lo = lane & 15, hf = lane >> 4;
    int qt = blockIdx.x, h = blockIdx.y, b = blockIdx.z;
    int q0 = qt * 128;
    size_t baseBS = (size_t)b * S;

    // Preload Q as A-fragments, held across the whole K loop
    Frag qa[2];
    {
        const __bf16* qp = Q + (baseBS + q0 + wave * 16 + lo) * DM + h * 64;
#pragma unroll
        for (int c = 0; c < 2; c++) {
            qa[c].f4[0] = *(const f32x4*)(qp + 32 * c + 8 * hf);
            qa[c].f4[1] = *(const f32x4*)(qp + 32 * c + 16 + 8 * hf);
        }
    }

    f32x8 oacc[4];
#pragma unroll
    for (int nt = 0; nt < 4; nt++)
#pragma unroll
        for (int r = 0; r < 8; r++) oacc[nt][r] = 0.f;
    float mrun[8], lrun[8];
#pragma unroll
    for (int r = 0; r < 8; r++) { mrun[r] = -3.0e38f; lrun[r] = 0.f; }

    for (int kt = 0; kt < 32; kt++) {
        // Cooperative tile loads: K row-major, V transposed
        {
            int kk = t >> 2, dseg = (t & 3) * 16;
            const __bf16* ksrc = Kg + (baseBS + kt * 64 + kk) * DM + h * 64 + dseg;
            *(f32x4*)&Ks[kk * 72 + dseg]     = *(const f32x4*)ksrc;
            *(f32x4*)&Ks[kk * 72 + dseg + 8] = *(const f32x4*)(ksrc + 8);

            union { f32x4 v[2]; unsigned short u[16]; } vl;
            const __bf16* vsrc = Vg + (baseBS + kt * 64 + kk) * DM + h * 64 + dseg;
            vl.v[0] = *(const f32x4*)vsrc;
            vl.v[1] = *(const f32x4*)(vsrc + 8);
            unsigned short* vtp = (unsigned short*)Vt;
#pragma unroll
            for (int j = 0; j < 16; j++) vtp[(dseg + j) * 72 + kk] = vl.u[j];
        }
        __syncthreads();

        // S = Q * K^T  (4 key sub-tiles x 2 d-chunks)
        f32x8 sacc[4];
#pragma unroll
        for (int nt = 0; nt < 4; nt++)
#pragma unroll
            for (int r = 0; r < 8; r++) sacc[nt][r] = 0.f;
#pragma unroll
        for (int c = 0; c < 2; c++) {
#pragma unroll
            for (int nt = 0; nt < 4; nt++) {
                Frag kb;
                const __bf16* kp = &Ks[(nt * 16 + lo) * 72 + 32 * c + 8 * hf];
                kb.f4[0] = *(const f32x4*)kp;
                kb.f4[1] = *(const f32x4*)(kp + 16);
                sacc[nt] = wmma_bf16(qa[c], kb, sacc[nt]);
            }
        }

        // scale + mask (mask depends only on key position)
        float pv[4][8];
#pragma unroll
        for (int nt = 0; nt < 4; nt++) {
            unsigned char mk = mask[baseBS + kt * 64 + nt * 16 + lo];
#pragma unroll
            for (int r = 0; r < 8; r++) {
                float sv = sacc[nt][r] * 0.125f;
                pv[nt][r] = mk ? -1.0e9f : sv;
            }
        }
        // online softmax: row max over the 64 keys of this tile
        float mnew[8];
#pragma unroll
        for (int r = 0; r < 8; r++)
            mnew[r] = fmaxf(fmaxf(pv[0][r], pv[1][r]), fmaxf(pv[2][r], pv[3][r]));
#pragma unroll
        for (int off = 1; off < 16; off <<= 1)
#pragma unroll
            for (int r = 0; r < 8; r++)
                mnew[r] = fmaxf(mnew[r], __shfl_xor(mnew[r], off));
        float alpha[8], rs[8];
#pragma unroll
        for (int r = 0; r < 8; r++) {
            mnew[r]  = fmaxf(mnew[r], mrun[r]);
            alpha[r] = __expf(mrun[r] - mnew[r]);
            mrun[r]  = mnew[r];
            rs[r]    = 0.f;
        }
#pragma unroll
        for (int nt = 0; nt < 4; nt++)
#pragma unroll
            for (int r = 0; r < 8; r++) {
                float e = __expf(pv[nt][r] - mnew[r]);
                pv[nt][r] = e; rs[r] += e;
            }
#pragma unroll
        for (int off = 1; off < 16; off <<= 1)
#pragma unroll
            for (int r = 0; r < 8; r++) rs[r] += __shfl_xor(rs[r], off);
#pragma unroll
        for (int r = 0; r < 8; r++) {
            lrun[r] = lrun[r] * alpha[r] + rs[r];
#pragma unroll
            for (int nt = 0; nt < 4; nt++) oacc[nt][r] *= alpha[r];
        }

        // stage P (C-layout -> A-layout via per-wave LDS region)
        {
            unsigned short* pw = (unsigned short*)&Ps[wave * 16 * 72];
#pragma unroll
            for (int nt = 0; nt < 4; nt++)
#pragma unroll
                for (int r = 0; r < 8; r++)
                    pw[(r + 8 * hf) * 72 + nt * 16 + lo] = f32_to_bf16(pv[nt][r]);
        }

        // O += P * V  (contraction over 64 keys, 4 d sub-tiles)
#pragma unroll
        for (int c = 0; c < 2; c++) {
            Frag pa;
            const __bf16* pp = &Ps[wave * 16 * 72 + lo * 72 + 32 * c];
            pa.f4[0] = *(const f32x4*)(pp + 8 * hf);
            pa.f4[1] = *(const f32x4*)(pp + 16 + 8 * hf);
#pragma unroll
            for (int nt = 0; nt < 4; nt++) {
                Frag vb;
                const __bf16* vp = &Vt[(nt * 16 + lo) * 72 + 32 * c + 8 * hf];
                vb.f4[0] = *(const f32x4*)vp;
                vb.f4[1] = *(const f32x4*)(vp + 16);
                oacc[nt] = wmma_bf16(pa, vb, oacc[nt]);
            }
        }
        __syncthreads();
    }

    // normalize and store bf16 attention output
#pragma unroll
    for (int nt = 0; nt < 4; nt++)
#pragma unroll
        for (int r = 0; r < 8; r++) {
            float o = oacc[nt][r] / lrun[r];
            size_t row = baseBS + q0 + wave * 16 + r + 8 * hf;
            O[row * DM + h * 64 + nt * 16 + lo] = f32_to_bf16(o);
        }
}

// ---------------------------------------------------------------------------
// Orchestration
// ---------------------------------------------------------------------------
extern "C" void kernel_launch(void* const* d_in, const int* in_sizes, int n_in,
                              void* d_out, int out_size, void* d_ws, size_t ws_size,
                              hipStream_t stream) {
    (void)in_sizes; (void)n_in; (void)out_size; (void)ws_size;
    const float* x       = (const float*)d_in[0];
    const float* w_q     = (const float*)d_in[1];
    const float* w_k     = (const float*)d_in[2];
    const float* w_v     = (const float*)d_in[3];
    const float* w_o     = (const float*)d_in[4];
    const float* l1_w    = (const float*)d_in[5];
    const float* l1_bias = (const float*)d_in[6];
    const float* l2_w    = (const float*)d_in[7];
    const float* l2_bias = (const float*)d_in[8];
    const float* norm1_a = (const float*)d_in[9];
    const float* norm1_b = (const float*)d_in[10];
    const float* norm2_a = (const float*)d_in[11];
    const float* norm2_b = (const float*)d_in[12];
    const float* normf_a = (const float*)d_in[13];
    const float* normf_b = (const float*)d_in[14];
    const unsigned char* mask = (const unsigned char*)d_in[15];

    const int M = 8192, D = 1024, F = 4096;
    const size_t MB = 1ull << 20;
    char* ws = (char*)d_ws;
    unsigned short* wq_b  = (unsigned short*)(ws + 0 * MB);    // 2 MB
    unsigned short* wk_b  = (unsigned short*)(ws + 2 * MB);    // 2 MB
    unsigned short* wv_b  = (unsigned short*)(ws + 4 * MB);    // 2 MB
    unsigned short* wo_b  = (unsigned short*)(ws + 6 * MB);    // 2 MB
    unsigned short* l1b16 = (unsigned short*)(ws + 8 * MB);    // 8 MB
    unsigned short* l2b16 = (unsigned short*)(ws + 16 * MB);   // 8 MB
    unsigned short* bufA  = (unsigned short*)(ws + 24 * MB);   // 16 MB (ln1 out, later ln2 out)
    unsigned short* qb    = (unsigned short*)(ws + 40 * MB);   // 16 MB
    unsigned short* kb    = (unsigned short*)(ws + 56 * MB);   // 16 MB
    unsigned short* vb    = (unsigned short*)(ws + 72 * MB);   // 16 MB
    unsigned short* attnb = (unsigned short*)(ws + 88 * MB);   // 16 MB
    float*          x1    = (float*)(ws + 104 * MB);           // 32 MB
    unsigned short* ffb   = qb;                                // 64 MB, reuses q/k/v/attn
    float*          x2    = (float*)(ws + 136 * MB);           // 32 MB  (total 168 MB)

    // 1) weights -> bf16
    auto cvt = [&](const float* s, unsigned short* d, int n) {
        int blocks = (n + 1023) / 1024;
        cvt_bf16_kernel<<<blocks, 256, 0, stream>>>(s, d, n);
    };
    cvt(w_q, wq_b, D * D);
    cvt(w_k, wk_b, D * D);
    cvt(w_v, wv_b, D * D);
    cvt(w_o, wo_b, D * D);
    cvt(l1_w, l1b16, F * D);
    cvt(l2_w, l2b16, D * F);

    dim3 gP(D / GEMM_BN, M / GEMM_BM);  // N=1024 GEMMs
    dim3 gF(F / GEMM_BN, M / GEMM_BM);  // N=4096 GEMM

    // 2) ln1(x) -> bf16
    ln_kernel<false, true><<<M, 256, 0, stream>>>(x, norm1_a, norm1_b, nullptr, bufA);

    // 3) Q,K,V projections (bf16 out)
    gemm_bf16_kernel<false, false, false, false, true><<<gP, 256, 0, stream>>>(
        (const __bf16*)bufA, (const __bf16*)wq_b, nullptr, nullptr, nullptr, qb, M, D, D);
    gemm_bf16_kernel<false, false, false, false, true><<<gP, 256, 0, stream>>>(
        (const __bf16*)bufA, (const __bf16*)wk_b, nullptr, nullptr, nullptr, kb, M, D, D);
    gemm_bf16_kernel<false, false, false, false, true><<<gP, 256, 0, stream>>>(
        (const __bf16*)bufA, (const __bf16*)wv_b, nullptr, nullptr, nullptr, vb, M, D, D);

    // 4) attention (flash, bf16 out)
    attn_kernel<<<dim3(16, 16, 4), 256, 0, stream>>>(
        (const __bf16*)qb, (const __bf16*)kb, (const __bf16*)vb, mask, attnb);

    // 5) output projection + residual(x) -> x1 (f32)
    gemm_bf16_kernel<false, false, true, true, false><<<gP, 256, 0, stream>>>(
        (const __bf16*)attnb, (const __bf16*)wo_b, nullptr, x, x1, nullptr, M, D, D);

    // 6) ln2(x1) -> bf16
    ln_kernel<false, true><<<M, 256, 0, stream>>>(x1, norm2_a, norm2_b, nullptr, bufA);

    // 7) FFN1: relu(h @ l1^T + b1) -> bf16
    gemm_bf16_kernel<true, true, false, false, true><<<gF, 256, 0, stream>>>(
        (const __bf16*)bufA, (const __bf16*)l1b16, l1_bias, nullptr, nullptr, ffb, M, F, D);

    // 8) FFN2: ff @ l2^T + b2 + residual(x1) -> x2 (f32)
    gemm_bf16_kernel<true, false, true, true, false><<<gP, 256, 0, stream>>>(
        (const __bf16*)ffb, (const __bf16*)l2b16, l2_bias, x1, x2, nullptr, M, D, F);

    // 9) final layer norm -> d_out (f32)
    ln_kernel<true, false><<<M, 256, 0, stream>>>(x2, normf_a, normf_b, (float*)d_out, nullptr);
}
